// MarkovRNN_36318243455103
// MI455X (gfx1250) — compile-verified, matrix operand order checked by reading, not run on hardware
//
#include <hip/hip_runtime.h>
#include <hip/hip_bf16.h>

// Problem constants
#define BB   4096
#define NX   512
#define NH   512
#define NY   256
#define KS   8      // Markov states

typedef __attribute__((ext_vector_type(16))) _Float16 v16h;
typedef __attribute__((ext_vector_type(8)))  _Float16 v8h;
typedef __attribute__((ext_vector_type(8)))  float    v8f;

// Branchless tanh: 1 - 2/(exp(2x)+1), via hardware v_exp_f32 / v_rcp_f32.
// Saturates correctly: x>>0 -> exp2->inf -> 1 ; x<<0 -> exp2->0 -> -1.
__device__ __forceinline__ float fast_tanh(float x) {
    // 2*log2(e) = 2.8853900817779268f
    float e = __builtin_amdgcn_exp2f(x * 2.8853900817779268f);
    return 1.0f - 2.0f * __builtin_amdgcn_rcpf(e + 1.0f);
}

// ---------------------------------------------------------------------------
// Convert kernels: build f16 operands with reduction-contiguous layouts.
// ---------------------------------------------------------------------------

// Abf[b][j] = j<512 ? x[b][j] : h[b][j-512]     (B x 1024)
__global__ __launch_bounds__(256) void conv_A(const float* __restrict__ x,
                                              const float* __restrict__ h,
                                              _Float16* __restrict__ Abf) {
    size_t idx = (size_t)blockIdx.x * blockDim.x + threadIdx.x;   // < B*1024
    int b = (int)(idx >> 10);
    int j = (int)(idx & 1023);
    float v = (j < NX) ? x[(size_t)b * NX + j] : h[(size_t)b * NH + (j - NX)];
    Abf[idx] = (_Float16)v;
}

// Wbf[k][n][j] = j<512 ? W_ih[k][n][j] : W_hh[k][n][j-512]   (K x 512 x 1024)
__global__ __launch_bounds__(256) void conv_W(const float* __restrict__ Wih,
                                              const float* __restrict__ Whh,
                                              _Float16* __restrict__ Wbf) {
    size_t idx = (size_t)blockIdx.x * blockDim.x + threadIdx.x;   // < K*512*1024
    int k = (int)(idx >> 19);
    int r = (int)(idx & 524287);
    int n = r >> 10;
    int j = r & 1023;
    float v = (j < NX) ? Wih[((size_t)k * NH + n) * NX + j]
                       : Whh[((size_t)k * NH + n) * NH + (j - NX)];
    Wbf[idx] = (_Float16)v;
}

// Woutbf[y][n] = W_out[y][n]   (256 x 512), reduction dim n contiguous already
__global__ __launch_bounds__(256) void conv_Wout(const float* __restrict__ Wo,
                                                 _Float16* __restrict__ Wobf) {
    size_t idx = (size_t)blockIdx.x * blockDim.x + threadIdx.x;   // < 256*512
    Wobf[idx] = (_Float16)Wo[idx];
}

// ---------------------------------------------------------------------------
// Unified TN GEMM: C[m][n] = sum_j A[m][j] * B[n][j], both f16, f32 accum.
//   Tile: 128(M) x 128(N), k-chunks of 32.  256 threads = 8 waves.
//   Wave (wm,wn) owns rows [wm*32,+32) x cols [wn*64,+64) = 2x4 WMMA subtiles.
//   mode doTanh=1: val=tanh(val); store f32 to outF[(m*ncols+n)*8+k] and
//                  f16 copy to outH[k*B*NH + m*NH + n] (for the next GEMM).
//   mode doTanh=0: store f32 to outF[(m*ncols+n)*8+k].
// ---------------------------------------------------------------------------
__global__ __launch_bounds__(256)
void markov_gemm(const _Float16* __restrict__ Abase, size_t aKStride, int lda,
                 const _Float16* __restrict__ Bbase, size_t bKStride, int ldb,
                 int rd, int ncols,
                 float* __restrict__ outF, _Float16* __restrict__ outH,
                 int doTanh) {
    constexpr int LDST = 40;                 // padded LDS row stride (f16 units)
    __shared__ _Float16 As[128 * LDST];
    __shared__ _Float16 Bs[128 * LDST];

    const int k       = blockIdx.z;
    const int rowBase = blockIdx.x * 128;
    const int colBase = blockIdx.y * 128;

    const _Float16* Ag = Abase + aKStride * k + (size_t)rowBase * lda;
    const _Float16* Bg = Bbase + bKStride * k + (size_t)colBase * ldb;

    const int t    = threadIdx.x;
    const int lane = t & 31;
    const int wave = t >> 5;
    const int wm   = wave >> 1;      // 0..3
    const int wn   = wave & 1;       // 0..1
    const int half = lane >> 4;      // 0..1  (lane group per ISA layout)
    const int l16  = lane & 15;

    v8f acc[2][4];
    const v8f z8 = {0.f, 0.f, 0.f, 0.f, 0.f, 0.f, 0.f, 0.f};
    #pragma unroll
    for (int i = 0; i < 2; ++i)
        #pragma unroll
        for (int j = 0; j < 4; ++j) acc[i][j] = z8;

    for (int j0 = 0; j0 < rd; j0 += 32) {
        if (j0 + 32 < rd) {
            // CDNA5 speculative prefetch of the next k-chunk (global_prefetch_b8)
            __builtin_prefetch(&Ag[(size_t)(t >> 1) * lda + j0 + 32], 0, 1);
            __builtin_prefetch(&Bg[(size_t)(t >> 1) * ldb + j0 + 32], 0, 1);
        }
        __syncthreads();
        // Stage A and B tiles: 128 rows x 32 f16 each = 512 chunks of 16B
        #pragma unroll
        for (int i = 0; i < 2; ++i) {
            int c  = t + i * 256;        // 0..511
            int r  = c >> 2;
            int cc = (c & 3) * 8;
            *(v8h*)&As[r * LDST + cc] = *(const v8h*)&Ag[(size_t)r * lda + j0 + cc];
            *(v8h*)&Bs[r * LDST + cc] = *(const v8h*)&Bg[(size_t)r * ldb + j0 + cc];
        }
        __syncthreads();

        // Fragments per ISA 16-bit A 16x32 layout:
        //   elems 0..7  -> K = 8*half + e ;  elems 8..15 -> K = 16 + 8*half + e
        v16h af[2], bfr[4];
        #pragma unroll
        for (int rt = 0; rt < 2; ++rt) {
            const _Float16* p = &As[(wm * 32 + rt * 16 + l16) * LDST];
            v8h lo = *(const v8h*)&p[8 * half];
            v8h hi = *(const v8h*)&p[16 + 8 * half];
            af[rt] = __builtin_shufflevector(lo, hi, 0,1,2,3,4,5,6,7,8,9,10,11,12,13,14,15);
        }
        #pragma unroll
        for (int ct = 0; ct < 4; ++ct) {
            const _Float16* p = &Bs[(wn * 64 + ct * 16 + l16) * LDST];
            v8h lo = *(const v8h*)&p[8 * half];
            v8h hi = *(const v8h*)&p[16 + 8 * half];
            bfr[ct] = __builtin_shufflevector(lo, hi, 0,1,2,3,4,5,6,7,8,9,10,11,12,13,14,15);
        }

        #pragma unroll
        for (int rt = 0; rt < 2; ++rt)
            #pragma unroll
            for (int ct = 0; ct < 4; ++ct)
                acc[rt][ct] = __builtin_amdgcn_wmma_f32_16x16x32_f16(
                    false, af[rt], false, bfr[ct], (short)0, acc[rt][ct],
                    false, false);
    }

    // Epilogue.  C/D layout: VGPR v holds M = v + 8*half, N = lane%16.
    #pragma unroll
    for (int rt = 0; rt < 2; ++rt) {
        #pragma unroll
        for (int ct = 0; ct < 4; ++ct) {
            int col = colBase + wn * 64 + ct * 16 + l16;
            #pragma unroll
            for (int v = 0; v < 8; ++v) {
                int row = rowBase + wm * 32 + rt * 16 + v + 8 * half;
                float val = acc[rt][ct][v];
                if (doTanh) {
                    val = fast_tanh(val);
                    outH[(size_t)k * BB * NH + (size_t)row * NH + col] = (_Float16)val;
                }
                outF[((size_t)row * ncols + col) * KS + k] = val;
            }
        }
    }
}

// ---------------------------------------------------------------------------
// Belief-weighted combine:
//   h_mix[b][n] = sum_k pi[b][k] * h_k[b][n][k]
//   out[b][y]   = sum_k pi[b][k] * out_k[b][y][k]
// ---------------------------------------------------------------------------
__global__ __launch_bounds__(256)
void combine(const float* __restrict__ hk, const float* __restrict__ outk,
             const float* __restrict__ pi,
             float* __restrict__ hmix, float* __restrict__ out) {
    const size_t N1 = (size_t)BB * NH;
    const size_t N2 = (size_t)BB * NY;
    size_t idx = (size_t)blockIdx.x * blockDim.x + threadIdx.x;
    if (idx < N1) {
        int b = (int)(idx >> 9);                 // NH = 512
        const float4 p0 = *(const float4*)&pi[(size_t)b * KS];
        const float4 p1 = *(const float4*)&pi[(size_t)b * KS + 4];
        const float4 v0 = *(const float4*)&hk[idx * KS];
        const float4 v1 = *(const float4*)&hk[idx * KS + 4];
        float s = p0.x * v0.x + p0.y * v0.y + p0.z * v0.z + p0.w * v0.w +
                  p1.x * v1.x + p1.y * v1.y + p1.z * v1.z + p1.w * v1.w;
        hmix[idx] = s;
    } else if (idx < N1 + N2) {
        size_t j = idx - N1;
        int b = (int)(j >> 8);                   // NY = 256
        const float4 p0 = *(const float4*)&pi[(size_t)b * KS];
        const float4 p1 = *(const float4*)&pi[(size_t)b * KS + 4];
        const float4 v0 = *(const float4*)&outk[j * KS];
        const float4 v1 = *(const float4*)&outk[j * KS + 4];
        float s = p0.x * v0.x + p0.y * v0.y + p0.z * v0.z + p0.w * v0.w +
                  p1.x * v1.x + p1.y * v1.y + p1.z * v1.z + p1.w * v1.w;
        out[j] = s;
    }
}

// ---------------------------------------------------------------------------
// Launch
// ---------------------------------------------------------------------------
extern "C" void kernel_launch(void* const* d_in, const int* in_sizes, int n_in,
                              void* d_out, int out_size, void* d_ws, size_t ws_size,
                              hipStream_t stream) {
    const float* x    = (const float*)d_in[0];
    const float* h    = (const float*)d_in[1];
    const float* pi   = (const float*)d_in[2];
    const float* Wih  = (const float*)d_in[3];
    const float* Whh  = (const float*)d_in[4];
    const float* Wout = (const float*)d_in[5];

    // Output tuple layout (floats): out | h_mix | out_k | h_k
    float* outP  = (float*)d_out;
    float* out   = outP;                                    // B*NY
    float* hmix  = outP + (size_t)BB * NY;                  // B*NH
    float* outk  = hmix + (size_t)BB * NH;                  // B*NY*K
    float* hk    = outk + (size_t)BB * NY * KS;             // B*NH*K

    // Workspace (f16): Abf | Wbf | Woutbf | hkbf  (~48.3 MB)
    _Float16* Abf    = (_Float16*)d_ws;
    _Float16* Wbf    = Abf + (size_t)BB * 1024;             // B x 1024
    _Float16* Wobf   = Wbf + (size_t)KS * NH * 1024;        // K x 512 x 1024
    _Float16* hkbf   = Wobf + (size_t)NY * NH;              // K x B x 512

    // 1) convert to f16, reduction-contiguous layouts
    conv_A   <<<((size_t)BB * 1024) / 256, 256, 0, stream>>>(x, h, Abf);
    conv_W   <<<((size_t)KS * NH * 1024) / 256, 256, 0, stream>>>(Wih, Whh, Wbf);
    conv_Wout<<<((size_t)NY * NH) / 256, 256, 0, stream>>>(Wout, Wobf);

    // 2) h_k = tanh([x|h] @ Wcat[k]^T)  -> hk (f32) + hkbf (f16)
    //    grid: (B/128, NH/128, K) = (32,4,8)
    markov_gemm<<<dim3(BB / 128, NH / 128, KS), 256, 0, stream>>>(
        Abf, (size_t)0, 1024,
        Wbf, (size_t)NH * 1024, 1024,
        /*rd=*/1024, /*ncols=*/NH,
        hk, hkbf, /*doTanh=*/1);

    // 3) out_k = h_k @ W_out^T    grid: (B/128, NY/128, K) = (32,2,8)
    markov_gemm<<<dim3(BB / 128, NY / 128, KS), 256, 0, stream>>>(
        hkbf, (size_t)BB * NH, NH,
        Wobf, (size_t)0, NH,
        /*rd=*/NH, /*ncols=*/NY,
        outk, nullptr, /*doTanh=*/0);

    // 4) belief-weighted combine -> h_mix, out
    size_t nComb = (size_t)BB * NH + (size_t)BB * NY;
    combine<<<(nComb + 255) / 256, 256, 0, stream>>>(hk, outk, pi, hmix, out);
}
